// DecoderBlock_RL_16183436772089
// MI455X (gfx1250) — matrix operand order repaired and checked by reading, not compiled, not run on hardware
//
#include <hip/hip_runtime.h>

#ifndef USE_ASYNC_LDS
#define USE_ASYNC_LDS 1
#endif

typedef _Float16 h4  __attribute__((ext_vector_type(4)));
typedef _Float16 v8h  __attribute__((ext_vector_type(8)));
typedef _Float16 v16h __attribute__((ext_vector_type(16)));
typedef float    v8f  __attribute__((ext_vector_type(8)));

__device__ __forceinline__ int imin(int a, int b) { return a < b ? a : b; }

// ---------------------------------------------------------------------------
// 16-byte global->LDS copy. Async (ASYNCcnt-tracked, overlapped with compute)
// on CDNA5; generic pointer low 32 bits are the wave LDS byte offset.
// ---------------------------------------------------------------------------
__device__ __forceinline__ void g2l_b128(_Float16* lds_ptr, const _Float16* g) {
#if USE_ASYNC_LDS
  unsigned loff = (unsigned)(unsigned long long)(void*)lds_ptr;
  asm volatile("global_load_async_to_lds_b128 %0, %1, off"
               :: "v"(loff), "v"(g) : "memory");
#else
  *(v8h*)lds_ptr = *(const v8h*)g;
#endif
}
__device__ __forceinline__ void async_wait_all() {
#if USE_ASYNC_LDS
  asm volatile("s_wait_asynccnt 0x0" ::: "memory");
#endif
}

// ---------------------------------------------------------------------------
// WMMA fragment loaders (CDNA5 16x16x32 f16 layouts, wave32)
// A 16x32: lane l holds row m=l%16; halves 0..7 -> K=kb..kb+7, 8..15 ->
//          K=16+kb..23+kb, kb = (l/16)*8.
// B 32x16: lane l holds col n=l%16; same K order as A (kb = (l/16)*8).
// C/D 16x16 f32: lane l, vgpr r -> (m = r + 8*(l/16), n = l%16).
// ---------------------------------------------------------------------------
__device__ __forceinline__ v16h frag_a(const _Float16* row, int kb) {
  union { v16h v; v8h h[2]; } u;
  u.h[0] = *(const v8h*)(row + kb);
  u.h[1] = *(const v8h*)(row + 16 + kb);
  return u.v;
}
__device__ __forceinline__ v16h frag_b(const _Float16* row, int kh) {
  union { v16h v; v8h h[2]; } u;
  u.h[0] = *(const v8h*)(row + (kh >> 1));
  u.h[1] = *(const v8h*)(row + 16 + (kh >> 1));
  return u.v;
}

// ---------------------------------------------------------------------------
// TN GEMM, all-f16 operands: C[z] = act(A[z/aDiv] @ B[z]^T * cScale + bias)
// A f16 [M][K] (lda), Bm f16 [N][K] (ldb).  Output f32 (Cf) or f16 (Ch).
// 128x128x32 tile, 8 waves (4m x 2n), async double-buffered LDS staging.
// Fully unrolled acc indexing -> accumulators stay in VGPRs (no scratch).
// ---------------------------------------------------------------------------
__global__ __launch_bounds__(256, 1) void gemm_f16_kernel(
    const _Float16* __restrict__ A, const _Float16* __restrict__ Bm,
    const float* __restrict__ bias, float* __restrict__ Cf,
    _Float16* __restrict__ Ch, int M, int N, int K,
    long lda, long ldb, long ldc,
    long aBatch, int aDiv, long bBatch, long cBatch,
    float cScale, int doRelu)
{
  __shared__ __align__(16) _Float16 As[2][128 * 40];
  __shared__ __align__(16) _Float16 Bs[2][128 * 40];

  int z = blockIdx.z;
  A += (long)(z / aDiv) * aBatch;
  Bm += (long)z * bBatch;
  if (Cf) Cf += (long)z * cBatch;
  if (Ch) Ch += (long)z * cBatch;

  int row0 = blockIdx.y * 128, col0 = blockIdx.x * 128;
  int tid = threadIdx.x, lane = tid & 31, wave = tid >> 5;
  int wm = wave & 3, wn = wave >> 2;
  int kb = (lane >> 4) << 3, kh = (lane >> 4) << 4, ln16 = lane & 15;
  int g8 = (lane >> 4) << 3;

  // Each tile = 128 rows x 32 halves = 512 x 16B chunks; 2 chunks/thread.
  int c0 = tid, c1 = tid + 256;
  int ar0 = c0 >> 2, ac0 = (c0 & 3) * 8;
  int ar1 = c1 >> 2, ac1 = (c1 & 3) * 8;
  const _Float16* Ag0 = A + (long)imin(row0 + ar0, M - 1) * lda + ac0;
  const _Float16* Ag1 = A + (long)imin(row0 + ar1, M - 1) * lda + ac1;
  const _Float16* Bg0 = Bm + (long)imin(col0 + ar0, N - 1) * ldb + ac0;
  const _Float16* Bg1 = Bm + (long)imin(col0 + ar1, N - 1) * ldb + ac1;

  v8f acc[2][4] = {};

  // prologue: stage k-tile 0 into buffer 0
  g2l_b128(&As[0][ar0 * 40 + ac0], Ag0);
  g2l_b128(&As[0][ar1 * 40 + ac1], Ag1);
  g2l_b128(&Bs[0][ar0 * 40 + ac0], Bg0);
  g2l_b128(&Bs[0][ar1 * 40 + ac1], Bg1);
  async_wait_all();
  __syncthreads();

  int buf = 0;
  for (int k0 = 0; k0 < K; k0 += 32) {
    if (k0 + 32 < K) {  // prefetch next k-tile into other buffer (overlapped)
      int nb = buf ^ 1, kn = k0 + 32;
      g2l_b128(&As[nb][ar0 * 40 + ac0], Ag0 + kn);
      g2l_b128(&As[nb][ar1 * 40 + ac1], Ag1 + kn);
      g2l_b128(&Bs[nb][ar0 * 40 + ac0], Bg0 + kn);
      g2l_b128(&Bs[nb][ar1 * 40 + ac1], Bg1 + kn);
    }
    v16h af[2], bf[4];
#pragma unroll
    for (int i = 0; i < 2; ++i)
      af[i] = frag_a(&As[buf][(wm * 32 + i * 16 + ln16) * 40], kb);
#pragma unroll
    for (int j = 0; j < 4; ++j)
      bf[j] = frag_b(&Bs[buf][(wn * 64 + j * 16 + ln16) * 40], kh);
#pragma unroll
    for (int i = 0; i < 2; ++i)
#pragma unroll
      for (int j = 0; j < 4; ++j)
        acc[i][j] = __builtin_amdgcn_wmma_f32_16x16x32_f16(
            false, af[i], false, bf[j], (short)0, acc[i][j], false, false);
    async_wait_all();
    __syncthreads();
    buf ^= 1;
  }

#pragma unroll
  for (int i = 0; i < 2; ++i) {
#pragma unroll
    for (int j = 0; j < 4; ++j) {
      int n = col0 + wn * 64 + j * 16 + ln16;
      if (n >= N) continue;
      float bv = bias ? bias[n] : 0.f;
      v8f av = acc[i][j];               // local copy, constant-index extracts
      long rowbase = row0 + wm * 32 + i * 16 + g8;
      if (Cf) {
#pragma unroll
        for (int r = 0; r < 8; ++r) {
          long m = rowbase + r;
          if (m < M) {
            float v = av[r] * cScale + bv;
            if (doRelu) v = v > 0.f ? v : 0.f;
            Cf[m * ldc + n] = v;
          }
        }
      } else {
#pragma unroll
        for (int r = 0; r < 8; ++r) {
          long m = rowbase + r;
          if (m < M) {
            float v = av[r] * cScale + bv;
            if (doRelu) v = v > 0.f ? v : 0.f;
            Ch[m * ldc + n] = (_Float16)v;
          }
        }
      }
    }
  }
}

// ---------------------------------------------------------------------------
// Flash attention, f16 in/out, head dim 64, model dim 1024, online softmax.
// mode 0: causal (key <= query). mode 1: key < vl[b].
// Grid: (Tq/128, B*H). Block 256 (8 waves x 16 query rows).
// ---------------------------------------------------------------------------
__global__ __launch_bounds__(256, 1) void flash_attn_kernel(
    const _Float16* __restrict__ Q, const _Float16* __restrict__ Kp,
    const _Float16* __restrict__ Vp, _Float16* __restrict__ O,
    int Tq, int Tk, int nheads, int mode, const int* __restrict__ vl,
    float scale)
{
  const int DM = 1024, DH = 64;
  __shared__ __align__(16) _Float16 Qs[128 * 72];
  __shared__ __align__(16) _Float16 Ks[64 * 72];
  __shared__ __align__(16) _Float16 Vs[64 * 72];   // transposed: Vs[d][key]
  __shared__ __align__(16) _Float16 Ps[128 * 72];
  __shared__ float rowM[128];
  __shared__ float rowS[128];

  int b = blockIdx.y / nheads;
  int h = blockIdx.y % nheads;
  int qb = blockIdx.x * 128;
  int tid = threadIdx.x, lane = tid & 31, wave = tid >> 5;
  int kb = (lane >> 4) << 3, kh = (lane >> 4) << 4;
  int ln16 = lane & 15, g8 = (lane >> 4) << 3;

  const _Float16* Qb = Q + ((long)b * Tq) * DM + h * DH;
  const _Float16* Kb = Kp + ((long)b * Tk) * DM + h * DH;
  const _Float16* Vb = Vp + ((long)b * Tk) * DM + h * DH;

  // stage Q tile: 128 rows x 64 halves = 1024 x 8-half chunks
#pragma unroll
  for (int p = 0; p < 4; ++p) {
    int c = tid + p * 256;
    int r = c >> 3, ch = (c & 7) * 8;
    g2l_b128(&Qs[r * 72 + ch], Qb + (long)(qb + r) * DM + ch);
  }
  if (tid < 128) { rowM[tid] = -1e30f; rowS[tid] = 0.f; }
  async_wait_all();
  __syncthreads();

  v16h qf[2];
  {
    const _Float16* qrow = Qs + (wave * 16 + ln16) * 72;
    qf[0] = frag_a(qrow, kb);
    qf[1] = frag_a(qrow + 32, kb);
  }
  v8f oacc[4] = {};

  int limit = (mode == 0) ? (qb + 128) : vl[b];
  if (limit > Tk) limit = Tk;
  int nkt = (limit + 63) / 64;

  for (int kt = 0; kt < nkt; ++kt) {
    int kb0 = kt * 64;
    // stage K tile (row copies) + V tile (scalar transpose)
#pragma unroll
    for (int p = 0; p < 2; ++p) {
      int c = tid + p * 256;
      int r = c >> 3, ch = (c & 7) * 8;
      int kr = imin(kb0 + r, Tk - 1);
      g2l_b128(&Ks[r * 72 + ch], Kb + (long)kr * DM + ch);
    }
#pragma unroll
    for (int p = 0; p < 16; ++p) {
      int e = tid + p * 256;
      int d = e & 63, r = e >> 6;
      int kr = imin(kb0 + r, Tk - 1);
      Vs[d * 72 + r] = Vb[(long)kr * DM + d];
    }
    async_wait_all();
    __syncthreads();

    v8f sacc[4];
#pragma unroll
    for (int j = 0; j < 4; ++j) {
      const _Float16* krow = Ks + (j * 16 + ln16) * 72;
      v8f sa = {};
      sa = __builtin_amdgcn_wmma_f32_16x16x32_f16(false, qf[0], false, frag_b(krow, kh),      (short)0, sa, false, false);
      sa = __builtin_amdgcn_wmma_f32_16x16x32_f16(false, qf[1], false, frag_b(krow + 32, kh), (short)0, sa, false, false);
      sacc[j] = sa;
    }

#pragma unroll
    for (int r = 0; r < 8; ++r) {
      int tr = wave * 16 + r + g8;
      int mrow = qb + tr;
      float mx = -1e30f;
#pragma unroll
      for (int j = 0; j < 4; ++j) {
        int ncol = kb0 + j * 16 + ln16;
        float v = sacc[j][r] * scale;
        bool ok = (ncol < Tk) && (mode == 0 ? (ncol <= mrow) : (ncol < limit));
        v = ok ? v : -1e6f;
        sacc[j][r] = v;
        mx = fmaxf(mx, v);
      }
#pragma unroll
      for (int m2 = 1; m2 < 16; m2 <<= 1) mx = fmaxf(mx, __shfl_xor(mx, m2, 32));
      float oldM = rowM[tr];
      float newM = fmaxf(oldM, mx);
      float sc = __expf(oldM - newM);
      float s = 0.f;
#pragma unroll
      for (int j = 0; j < 4; ++j) {
        float p = __expf(sacc[j][r] - newM);
        sacc[j][r] = p;
        s += p;
      }
#pragma unroll
      for (int m2 = 1; m2 < 16; m2 <<= 1) s += __shfl_xor(s, m2, 32);
      if (ln16 == 0) { rowM[tr] = newM; rowS[tr] = rowS[tr] * sc + s; }
#pragma unroll
      for (int c = 0; c < 4; ++c) oacc[c][r] *= sc;
#pragma unroll
      for (int j = 0; j < 4; ++j)
        Ps[tr * 72 + j * 16 + ln16] = (_Float16)sacc[j][r];
    }
    __syncthreads();

    {
      const _Float16* prow = Ps + (wave * 16 + ln16) * 72;
      v16h pf0 = frag_a(prow, kb);
      v16h pf1 = frag_a(prow + 32, kb);
#pragma unroll
      for (int c = 0; c < 4; ++c) {
        const _Float16* vrow = Vs + (c * 16 + ln16) * 72;
        oacc[c] = __builtin_amdgcn_wmma_f32_16x16x32_f16(false, pf0, false, frag_b(vrow, kh),      (short)0, oacc[c], false, false);
        oacc[c] = __builtin_amdgcn_wmma_f32_16x16x32_f16(false, pf1, false, frag_b(vrow + 32, kh), (short)0, oacc[c], false, false);
      }
    }
    __syncthreads();
  }

#pragma unroll
  for (int r = 0; r < 8; ++r) {
    int tr = wave * 16 + r + g8;
    float inv = 1.f / rowS[tr];
#pragma unroll
    for (int c = 0; c < 4; ++c)
      O[((long)b * Tq + qb + tr) * DM + h * DH + c * 16 + ln16] =
          (_Float16)(oacc[c][r] * inv);
  }
}

// ---------------------------------------------------------------------------
// f32 -> f16 convert (n multiple of 1024; 4 elems/thread)
// ---------------------------------------------------------------------------
__global__ __launch_bounds__(256) void cvt_f16_kernel(
    const float* __restrict__ in, _Float16* __restrict__ out, long n)
{
  long i = ((long)blockIdx.x * 256 + threadIdx.x) * 4;
  if (i >= n) return;
  float4 v = *(const float4*)(in + i);
  h4 o = {(_Float16)v.x, (_Float16)v.y, (_Float16)v.z, (_Float16)v.w};
  *(h4*)(out + i) = o;
}

// f32 [K][N] -> f16 [N][K] (transpose + convert), 32x32 LDS tiles
__global__ __launch_bounds__(256) void cvtT_f16_kernel(
    const float* __restrict__ in, _Float16* __restrict__ out, int K, int N)
{
  __shared__ float tile[32][33];
  int k0 = blockIdx.y * 32, n0 = blockIdx.x * 32;
  int tx = threadIdx.x & 31, ty = threadIdx.x >> 5;
  for (int i = ty; i < 32; i += 8) {
    int k = k0 + i, n = n0 + tx;
    tile[i][tx] = (k < K && n < N) ? in[(long)k * N + n] : 0.f;
  }
  __syncthreads();
  for (int i = ty; i < 32; i += 8) {
    int n = n0 + i, k = k0 + tx;
    if (n < N && k < K) out[(long)n * K + k] = (_Float16)tile[tx][i];
  }
}

// ---------------------------------------------------------------------------
// Stat attention: scores qs.ks/32, mask by stat_valid, top-8, softmax -> sw.
// ---------------------------------------------------------------------------
__global__ __launch_bounds__(256) void stat_softmax_kernel(
    const float* __restrict__ qs, const float* __restrict__ ksf,
    const int* __restrict__ vlen, float* __restrict__ sw,
    int T_, int S_, int D_)
{
  __shared__ float sc[16];
  long bt = blockIdx.x;
  int b = bt / T_;
  int tid = threadIdx.x;
  int s = tid >> 4, ln = tid & 15;
  const float* q = qs + bt * D_;
  const float* k = ksf + ((long)b * S_ + s) * D_;
  float p = 0.f;
  for (int d = ln; d < D_; d += 16) p += q[d] * k[d];
  for (int m2 = 1; m2 < 16; m2 <<= 1) p += __shfl_xor(p, m2, 32);
  if (ln == 0) sc[s] = p * (1.f / 32.f);
  __syncthreads();
  if (tid < 16) {
    int vlv = vlen[b];
    float v = (tid < vlv) ? sc[tid] : -1e6f;
    int rank = 0;
    for (int j = 0; j < 16; ++j) {
      float vj = (j < vlv) ? sc[j] : -1e6f;
      if (vj > v || (vj == v && j < tid)) rank++;
    }
    float kept = (rank < 8) ? v : -1e6f;
    float mx = kept;
    for (int m2 = 1; m2 < 16; m2 <<= 1) mx = fmaxf(mx, __shfl_xor(mx, m2, 32));
    float e = __expf(kept - mx);
    float sum = e;
    for (int m2 = 1; m2 < 16; m2 <<= 1) sum += __shfl_xor(sum, m2, 32);
    sw[bt * S_ + tid] = e / sum;
  }
}

// ---------------------------------------------------------------------------
// Token top-16 + softmax per (b,t,s) row of N=256 scores. One wave per row.
// ---------------------------------------------------------------------------
__global__ __launch_bounds__(256) void token_topk_kernel(
    const float* __restrict__ tsc, float* __restrict__ twv,
    int* __restrict__ twi, long nRows, int N_)
{
  int wave = threadIdx.x >> 5;
  int lane = threadIdx.x & 31;
  long row = (long)blockIdx.x * 8 + wave;
  if (row >= nRows) return;
  const float* src = tsc + row * N_;
  int base = lane * 8;
  float v[8];
#pragma unroll
  for (int i = 0; i < 8; ++i) v[i] = src[base + i];
  float vals[16]; int idxs[16];
#pragma unroll
  for (int t = 0; t < 16; ++t) {
    float lm = -1e30f; int li = 0x7fffffff;
#pragma unroll
    for (int i = 0; i < 8; ++i)
      if (v[i] > lm) { lm = v[i]; li = base + i; }
#pragma unroll
    for (int m2 = 1; m2 < 32; m2 <<= 1) {
      float om = __shfl_xor(lm, m2, 32);
      int oi = __shfl_xor(li, m2, 32);
      if (om > lm || (om == lm && oi < li)) { lm = om; li = oi; }
    }
    vals[t] = lm; idxs[t] = li;
    int sl = li - base;
#pragma unroll
    for (int i = 0; i < 8; ++i)
      if (i == sl) v[i] = -1e30f;
  }
  float mx = vals[0];
  float sum = 0.f;
#pragma unroll
  for (int t = 0; t < 16; ++t) sum += __expf(vals[t] - mx);
  float inv = 1.f / sum;
  if (lane < 16) {
    float ov = 0.f; int oi = 0;
#pragma unroll
    for (int t = 0; t < 16; ++t)
      if (t == lane) { ov = vals[t]; oi = idxs[t]; }
    twv[row * 16 + lane] = __expf(ov - mx) * inv;
    twi[row * 16 + lane] = oi;
  }
}

// ---------------------------------------------------------------------------
// y2s_pre = sum_s sw * sum_j tw * vv[b,s,idx_j,:]  (sparse gather, f16 out)
// ---------------------------------------------------------------------------
__global__ __launch_bounds__(256) void gather_y2s_kernel(
    const float* __restrict__ sw, const float* __restrict__ twv,
    const int* __restrict__ twi, const float* __restrict__ vv,
    _Float16* __restrict__ out, int T_, int S_, int N_, int D_)
{
  long bt = blockIdx.x;
  int b = bt / T_;
  int d0 = threadIdx.x * 4;
  float4 acc = {0.f, 0.f, 0.f, 0.f};
  for (int s = 0; s < S_; ++s) {
    float ws = sw[bt * S_ + s];
    if (ws == 0.f) continue;
    const float* tv = twv + (bt * S_ + s) * 16;
    const int* ti = twi + (bt * S_ + s) * 16;
    const float* vbase = vv + ((long)b * S_ + s) * N_ * D_;
#pragma unroll 4
    for (int j = 0; j < 16; ++j) {
      float w = ws * tv[j];
      const float4 r = *(const float4*)(vbase + (long)ti[j] * D_ + d0);
      acc.x += w * r.x; acc.y += w * r.y; acc.z += w * r.z; acc.w += w * r.w;
    }
  }
  h4 o = {(_Float16)acc.x, (_Float16)acc.y, (_Float16)acc.z, (_Float16)acc.w};
  *(h4*)(out + bt * D_ + d0) = o;
}

// ---------------------------------------------------------------------------
// gate = sigmoid([y2s,y2e]@Wg); y2 = g*y2s + (1-g)*y2e
// ---------------------------------------------------------------------------
__global__ __launch_bounds__(256) void gate_kernel(
    const float* __restrict__ y2s, const float* __restrict__ y2e,
    const float* __restrict__ Wg, float* __restrict__ y2, int D_)
{
  __shared__ float red[8];
  long bt = blockIdx.x;
  int tid = threadIdx.x;
  const float* a = y2s + bt * D_;
  const float* e = y2e + bt * D_;
  float p = 0.f;
  for (int d = tid; d < D_; d += 256) p += a[d] * Wg[d] + e[d] * Wg[D_ + d];
  for (int m2 = 1; m2 < 32; m2 <<= 1) p += __shfl_xor(p, m2, 32);
  if ((tid & 31) == 0) red[tid >> 5] = p;
  __syncthreads();
  float tot = 0.f;
  for (int i = 0; i < 8; ++i) tot += red[i];
  float g = 1.f / (1.f + __expf(-tot));
  float* o = y2 + bt * D_;
  for (int d = tid; d < D_; d += 256) o[d] = g * a[d] + (1.f - g) * e[d];
}

// ---------------------------------------------------------------------------
// out = LN(a + alpha*b) * gamma + beta   (D = 1024, block 256)
// ---------------------------------------------------------------------------
__global__ __launch_bounds__(256) void add_ln_kernel(
    const float* __restrict__ a, const float* __restrict__ bsrc, float alpha,
    const float* __restrict__ gamma, const float* __restrict__ beta,
    float* __restrict__ out, int D_)
{
  __shared__ float rs[8], rq[8];
  long row = blockIdx.x;
  int tid = threadIdx.x;
  const float* ap = a + row * D_;
  const float* bp = bsrc + row * D_;
  float s = 0.f, q = 0.f, vloc[4];
#pragma unroll
  for (int i = 0; i < 4; ++i) {
    int d = tid + i * 256;
    float v = ap[d] + alpha * bp[d];
    vloc[i] = v; s += v; q += v * v;
  }
#pragma unroll
  for (int m2 = 1; m2 < 32; m2 <<= 1) {
    s += __shfl_xor(s, m2, 32);
    q += __shfl_xor(q, m2, 32);
  }
  if ((tid & 31) == 0) { rs[tid >> 5] = s; rq[tid >> 5] = q; }
  __syncthreads();
  s = 0.f; q = 0.f;
#pragma unroll
  for (int i = 0; i < 8; ++i) { s += rs[i]; q += rq[i]; }
  float mean = s / D_;
  float var = q / D_ - mean * mean;
  float inv = rsqrtf(var + 1e-5f);
#pragma unroll
  for (int i = 0; i < 4; ++i) {
    int d = tid + i * 256;
    out[row * D_ + d] = (vloc[i] - mean) * inv * gamma[d] + beta[d];
  }
}

// yi = concat(y, broadcast(intent)), f16 output
__global__ __launch_bounds__(256) void build_yi_kernel(
    const float* __restrict__ y, const float* __restrict__ intent,
    _Float16* __restrict__ yi, int T_, int D_, int DI_)
{
  long bt = blockIdx.x;
  int b = bt / T_;
  int DT = D_ + DI_;
  for (int c = threadIdx.x; c < DT; c += 256) {
    float v = (c < D_) ? y[bt * D_ + c] : intent[(long)b * DI_ + (c - D_)];
    yi[bt * (long)DT + c] = (_Float16)v;
  }
}

// ---------------------------------------------------------------------------
// Host orchestration
// ---------------------------------------------------------------------------
extern "C" void kernel_launch(void* const* d_in, const int* in_sizes, int n_in,
                              void* d_out, int out_size, void* d_ws, size_t ws_size,
                              hipStream_t stream) {
  constexpr int B = 4, T = 1024, D = 1024, DI = 256, DFF = 4096, H = 16;
  constexpr int S = 16, N = 256, E = 1024;
  constexpr long BT = (long)B * T;

  const float* x         = (const float*)d_in[0];
  const float* stat_enc  = (const float*)d_in[1];
  const float* exem      = (const float*)d_in[2];
  const float* stat_feat = (const float*)d_in[3];
  const float* intent    = (const float*)d_in[4];
  const float* Wq1 = (const float*)d_in[5],  *Wk1 = (const float*)d_in[6];
  const float* Wv1 = (const float*)d_in[7],  *Wo1 = (const float*)d_in[8];
  const float* Wq_stat  = (const float*)d_in[9],  *Wq_token = (const float*)d_in[10];
  const float* Wk_stat  = (const float*)d_in[11], *Wk_token = (const float*)d_in[12];
  const float* Wv_sel   = (const float*)d_in[13], *Wo_sel   = (const float*)d_in[14];
  const float* Wq2 = (const float*)d_in[15], *Wk2 = (const float*)d_in[16];
  const float* Wv2 = (const float*)d_in[17], *Wo2 = (const float*)d_in[18];
  const float* Wg  = (const float*)d_in[19];
  const float* W1  = (const float*)d_in[20], *b1 = (const float*)d_in[21];
  const float* W2  = (const float*)d_in[22], *b2 = (const float*)d_in[23];
  const float* g1  = (const float*)d_in[24], *be1 = (const float*)d_in[25];
  const float* g2  = (const float*)d_in[26], *be2 = (const float*)d_in[27];
  const float* g3  = (const float*)d_in[28], *be3 = (const float*)d_in[29];
  const int* stat_vl = (const int*)d_in[30];
  const int* ex_vl   = (const int*)d_in[31];
  float* out = (float*)d_out;

  // ---- workspace carve-out (byte allocator, 256B aligned) ----
  char* base = (char*)d_ws;
  size_t off = 0;
  auto allocB = [&](size_t bytes) {
    void* p = base + off;
    off = (off + bytes + 255) & ~(size_t)255;
    return p;
  };
  // f32 buffers
  float* fA   = (float*)allocB(BT * D * 4);            // x2 -> y2s -> ffn_out
  float* fB   = (float*)allocB(BT * D * 4);            // qs -> y2e
  float* fC   = (float*)allocB(BT * D * 4);            // y2 (gated)
  float* fY   = (float*)allocB(BT * D * 4);            // y
  float* fZ   = (float*)allocB(BT * D * 4);            // z
  float* fVV  = (float*)allocB((long)B * S * N * D * 4);  // vv
  float* fTSC = (float*)allocB((long)B * S * (long)T * N * 4);  // token scores
  float* fKS  = (float*)allocB((long)B * S * D * 4);   // stat keys
  float* fSW  = (float*)allocB(BT * S * 4);            // stat weights
  float* fTWV = (float*)allocB(BT * S * 16 * 4);       // topk probs
  int*   iTWI = (int*)allocB(BT * S * 16 * 4);         // topk indices
  // f16 activations
  _Float16* hX   = (_Float16*)allocB(BT * D * 2);      // x -> exemplar
  _Float16* hQ   = (_Float16*)allocB(BT * D * 2);      // q1 -> q2
  _Float16* hK   = (_Float16*)allocB(BT * D * 2);      // k1 -> k2
  _Float16* hV   = (_Float16*)allocB(BT * D * 2);      // v1 -> v2
  _Float16* hATT = (_Float16*)allocB(BT * D * 2);      // att1 -> att2
  _Float16* hYI  = (_Float16*)allocB(BT * (D + DI) * 2);
  _Float16* hQT  = (_Float16*)allocB(BT * D * 2);      // qt
  _Float16* hSE  = (_Float16*)allocB((long)B * S * N * D * 2);  // stat_enc -> ffn hidden
  _Float16* hKT  = (_Float16*)allocB((long)B * S * N * D * 2);  // kt
  _Float16* hY2P = (_Float16*)allocB(BT * D * 2);      // y2s_pre -> z(f16)
  _Float16* hSF  = (_Float16*)allocB((long)B * S * D * 2);      // stat_feat
  // f16 transposed weights [N][K]
  auto allocW = [&](long k, long n) { return (_Float16*)allocB(k * n * 2); };
  _Float16 *hWq1 = allocW(D, D), *hWk1 = allocW(D, D), *hWv1 = allocW(D, D), *hWo1 = allocW(D, D);
  _Float16 *hWqs = allocW(D + DI, D), *hWqt = allocW(D + DI, D);
  _Float16 *hWks = allocW(D, D), *hWkt = allocW(D, D), *hWvs = allocW(D, D), *hWos = allocW(D, D);
  _Float16 *hWq2 = allocW(D + DI, D), *hWk2 = allocW(D, D), *hWv2 = allocW(D, D), *hWo2 = allocW(D, D);
  _Float16 *hW1 = allocW(D, DFF), *hW2 = allocW(DFF, D);

  auto cvt = [&](const float* in, _Float16* o, long n) {
    cvt_f16_kernel<<<(unsigned)(n / 1024), 256, 0, stream>>>(in, o, n);
  };
  auto cvtT = [&](const float* in, _Float16* o, int K, int Nn) {
    cvtT_f16_kernel<<<dim3((Nn + 31) / 32, (K + 31) / 32), 256, 0, stream>>>(in, o, K, Nn);
  };
  auto gemm = [&](const _Float16* Ap, const _Float16* Bp, const float* biasp,
                  float* Cfp, _Float16* Chp, int M, int Nn, int K,
                  long lda, long ldb, long ldc,
                  long aB, int aDiv, long bB, long cB, int batch,
                  float cs, int relu) {
    dim3 grid((Nn + 127) / 128, (M + 127) / 128, batch);
    gemm_f16_kernel<<<grid, 256, 0, stream>>>(Ap, Bp, biasp, Cfp, Chp, M, Nn, K,
        lda, ldb, ldc, aB, aDiv, bB, cB, cs, relu);
  };

  // ---- one-time conversions ----
  cvtT(Wq1, hWq1, D, D);  cvtT(Wk1, hWk1, D, D);  cvtT(Wv1, hWv1, D, D);  cvtT(Wo1, hWo1, D, D);
  cvtT(Wq_stat, hWqs, D + DI, D);  cvtT(Wq_token, hWqt, D + DI, D);
  cvtT(Wk_stat, hWks, D, D);  cvtT(Wk_token, hWkt, D, D);
  cvtT(Wv_sel, hWvs, D, D);  cvtT(Wo_sel, hWos, D, D);
  cvtT(Wq2, hWq2, D + DI, D);  cvtT(Wk2, hWk2, D, D);  cvtT(Wv2, hWv2, D, D);  cvtT(Wo2, hWo2, D, D);
  cvtT(W1, hW1, D, DFF);  cvtT(W2, hW2, DFF, D);
  cvt(x, hX, BT * D);
  cvt(stat_feat, hSF, (long)B * S * D);
  cvt(stat_enc, hSE, (long)B * S * N * D);

  // ---- self-attention block ----
  gemm(hX, hWq1, nullptr, nullptr, hQ, BT, D, D, D, D, D, 0, 1, 0, 0, 1, 1.f, 0);
  gemm(hX, hWk1, nullptr, nullptr, hK, BT, D, D, D, D, D, 0, 1, 0, 0, 1, 1.f, 0);
  gemm(hX, hWv1, nullptr, nullptr, hV, BT, D, D, D, D, D, 0, 1, 0, 0, 1, 1.f, 0);
  flash_attn_kernel<<<dim3(T / 128, B * H), 256, 0, stream>>>(
      hQ, hK, hV, hATT, T, T, H, 0, nullptr, 0.125f);
  gemm(hATT, hWo1, nullptr, fA, nullptr, BT, D, D, D, D, D, 0, 1, 0, 0, 1, 1.f, 0);
  add_ln_kernel<<<BT, 256, 0, stream>>>(x, fA, 1.f, g1, be1, fY, D);
  build_yi_kernel<<<BT, 256, 0, stream>>>(fY, intent, hYI, T, D, DI);

  // ---- selective hierarchical top-k attention ----
  gemm(hYI, hWqs, nullptr, fB, nullptr, BT, D, D + DI, D + DI, D + DI, D, 0, 1, 0, 0, 1, 1.f, 0);
  gemm(hYI, hWqt, nullptr, nullptr, hQT, BT, D, D + DI, D + DI, D + DI, D, 0, 1, 0, 0, 1, 1.f, 0);
  gemm(hSF, hWks, nullptr, fKS, nullptr, B * S, D, D, D, D, D, 0, 1, 0, 0, 1, 1.f, 0);
  gemm(hSE, hWkt, nullptr, nullptr, hKT, B * S * N, D, D, D, D, D, 0, 1, 0, 0, 1, 1.f, 0);
  gemm(hSE, hWvs, nullptr, fVV, nullptr, B * S * N, D, D, D, D, D, 0, 1, 0, 0, 1, 1.f, 0);
  stat_softmax_kernel<<<BT, 256, 0, stream>>>(fB, fKS, stat_vl, fSW, T, S, D);
  // tsc[z=b*S+s] = qt[b] @ kt[b,s]^T / 32   (kt rows are already [n][k])
  gemm(hQT, hKT, nullptr, fTSC, nullptr, T, N, D, D, D, N,
       (long)T * D, S, (long)N * D, (long)T * N, B * S, 1.f / 32.f, 0);
  token_topk_kernel<<<(unsigned)((B * (long)T * S) / 8), 256, 0, stream>>>(
      fTSC, fTWV, iTWI, (long)B * T * S, N);
  gather_y2s_kernel<<<BT, 256, 0, stream>>>(fSW, fTWV, iTWI, fVV, hY2P, T, S, N, D);
  gemm(hY2P, hWos, nullptr, fA, nullptr, BT, D, D, D, D, D, 0, 1, 0, 0, 1, 1.f, 0);

  // ---- exemplar cross-attention ----
  cvt(exem, hX, (long)B * E * D);   // reuse hX (x f16 dead)
  gemm(hYI, hWq2, nullptr, nullptr, hQ, BT, D, D + DI, D + DI, D + DI, D, 0, 1, 0, 0, 1, 1.f, 0);
  gemm(hX, hWk2, nullptr, nullptr, hK, B * E, D, D, D, D, D, 0, 1, 0, 0, 1, 1.f, 0);
  gemm(hX, hWv2, nullptr, nullptr, hV, B * E, D, D, D, D, D, 0, 1, 0, 0, 1, 1.f, 0);
  flash_attn_kernel<<<dim3(T / 128, B * H), 256, 0, stream>>>(
      hQ, hK, hV, hATT, T, E, H, 1, ex_vl, 0.125f);
  gemm(hATT, hWo2, nullptr, fB, nullptr, BT, D, D, D, D, D, 0, 1, 0, 0, 1, 1.f, 0);

  // ---- gate, residual, FFN ----
  gate_kernel<<<BT, 256, 0, stream>>>(fA, fB, Wg, fC, D);
  add_ln_kernel<<<BT, 256, 0, stream>>>(fY, fC, 2.f, g2, be2, fZ, D);
  cvt(fZ, hY2P, BT * D);            // reuse hY2P as z(f16)
  gemm(hY2P, hW1, b1, nullptr, hSE, BT, DFF, D, D, D, DFF, 0, 1, 0, 0, 1, 1.f, 1);  // hSE reused as hidden
  gemm(hSE, hW2, b2, fA, nullptr, BT, D, DFF, DFF, DFF, D, 0, 1, 0, 0, 1, 1.f, 0);
  add_ln_kernel<<<BT, 256, 0, stream>>>(fZ, fA, 1.f, g3, be3, out, D);

  (void)in_sizes; (void)n_in; (void)out_size; (void)ws_size;
}
